// ScaledDotProductSelfAttentionBlock_53025666236870
// MI455X (gfx1250) — matrix operand order repaired and checked by reading, not compiled, not run on hardware
//
#include <hip/hip_runtime.h>

#define N_TOK 8192
#define E_DIM 1024
#define SCALE 0.03125f   // 1/sqrt(1024)

typedef __attribute__((ext_vector_type(16))) __bf16 v16bf;
typedef __attribute__((ext_vector_type(8)))  __bf16 v8bf;
typedef __attribute__((ext_vector_type(8)))  float  v8f;

union BF16x16 { v16bf v; __bf16 e[16]; };
union BF16x8  { v8bf  v; __bf16 e[8];  };
union F32x8   { v8f   v; float  e[8];  };

__device__ __forceinline__ v8f wmma_bf16(v16bf a, v16bf b, v8f c) {
  // D = A(16x32 bf16) * B(32x16 bf16) + C(16x16 f32)
  return __builtin_amdgcn_wmma_f32_16x16x32_bf16(false, a, false, b, (short)0, c,
                                                 false, false);
}

// ---- CDNA5 async global->LDS copy (ASYNCcnt-tracked, no VGPR round trip) ----
// VDST = per-lane LDS byte offset, VADDR = per-lane 64-bit global address.
__device__ __forceinline__ unsigned lds_off(const void* p) {
  // generic pointers to LDS carry the LDS byte offset in the low 32 bits
  return (unsigned)(uintptr_t)p;
}
__device__ __forceinline__ void async_load_b128(unsigned dst_lds, const void* src) {
  asm volatile("global_load_async_to_lds_b128 %0, %1, off"
               :: "v"(dst_lds), "v"((unsigned long long)(uintptr_t)src)
               : "memory");
}
__device__ __forceinline__ void wait_async0() {
  asm volatile("s_wait_asynccnt 0x0" ::: "memory");
}

// ---- Fragment loaders (wave32; contiguous K per lane half, same order for A and B) ----
// A 16x32: lane = g*16+m holds row m; elems 0..15 -> K = g*16 .. g*16+15 (contiguous)
__device__ __forceinline__ v16bf load_a_bf16(const __bf16* base, int ld, int row0,
                                             int k0, int g, int hl) {
  const __bf16* p = base + (size_t)(row0 + hl) * ld + k0 + g * 16;
  BF16x8 lo, hi;
  lo.v = *(const v8bf*)p;
  hi.v = *(const v8bf*)(p + 8);
  BF16x16 A;
#pragma unroll
  for (int i = 0; i < 8; ++i) { A.e[i] = lo.e[i]; A.e[i + 8] = hi.e[i]; }
  return A.v;
}

// B 32x16 with B[k,n] = M[n0+n, k0+k] (M row-major, K contiguous):
// lane = g*16+n holds col n; elems 0..15 -> K = g*16 .. g*16+15 (contiguous)
__device__ __forceinline__ v16bf load_b_bf16(const __bf16* base, int ld, int n0,
                                             int k0, int g, int hl) {
  const __bf16* p = base + (size_t)(n0 + hl) * ld + k0 + g * 16;
  BF16x8 lo, hi;
  lo.v = *(const v8bf*)p;
  hi.v = *(const v8bf*)(p + 8);
  BF16x16 B;
#pragma unroll
  for (int i = 0; i < 8; ++i) { B.e[i] = lo.e[i]; B.e[i + 8] = hi.e[i]; }
  return B.v;
}

__device__ __forceinline__ v16bf load_a_f32(const float* base, int ld, int row0,
                                            int k0, int g, int hl) {
  const float* p = base + (size_t)(row0 + hl) * ld + k0 + g * 16;
  BF16x16 A;
#pragma unroll
  for (int i = 0; i < 8; ++i) A.e[i] = (__bf16)p[i];
#pragma unroll
  for (int i = 0; i < 8; ++i) A.e[8 + i] = (__bf16)p[8 + i];
  return A.v;
}

__device__ __forceinline__ v16bf load_b_f32(const float* base, int ld, int n0,
                                            int k0, int g, int hl) {
  const float* p = base + (size_t)(n0 + hl) * ld + k0 + g * 16;
  BF16x16 B;
#pragma unroll
  for (int i = 0; i < 16; ++i) B.e[i] = (__bf16)p[i];
  return B.v;
}

// =====================================================================
// Kernel 1: q/k = x @ W^T + b (bf16 out), vT[e,n] = (x @ Wv^T + bv)^T
// Block: 256 thr (8 waves), tile 128 rows x 256 cols, wave tile 64x64.
// grid = (N/128, E/256, 3).  f32->bf16 cvts co-execute under the WMMA pipe.
// =====================================================================
__global__ __launch_bounds__(256) void sdpa_qkv_proj_kernel(
    const float* __restrict__ x,
    const float* __restrict__ Wq, const float* __restrict__ bq,
    const float* __restrict__ Wk, const float* __restrict__ bk,
    const float* __restrict__ Wv, const float* __restrict__ bv,
    __bf16* __restrict__ qo, __bf16* __restrict__ ko, __bf16* __restrict__ vTo) {
  const int lane = threadIdx.x & 31;
  const int wave = threadIdx.x >> 5;
  const int g = lane >> 4, hl = lane & 15;
  const int row0 = blockIdx.x * 128 + (wave >> 2) * 64;
  const int col0 = blockIdx.y * 256 + (wave & 3) * 64;
  const int proj = blockIdx.z;
  const float* W    = (proj == 0) ? Wq : ((proj == 1) ? Wk : Wv);
  const float* bias = (proj == 0) ? bq : ((proj == 1) ? bk : bv);

  F32x8 acc[4][4];
#pragma unroll
  for (int i = 0; i < 4; ++i)
#pragma unroll
    for (int j = 0; j < 4; ++j)
#pragma unroll
      for (int r = 0; r < 8; ++r) acc[i][j].e[r] = 0.0f;

  for (int k0 = 0; k0 < E_DIM; k0 += 32) {
    v16bf afr[4], bfr[4];
#pragma unroll
    for (int i = 0; i < 4; ++i) afr[i] = load_a_f32(x, E_DIM, row0 + i * 16, k0, g, hl);
#pragma unroll
    for (int j = 0; j < 4; ++j) bfr[j] = load_b_f32(W, E_DIM, col0 + j * 16, k0, g, hl);
#pragma unroll
    for (int i = 0; i < 4; ++i)
#pragma unroll
      for (int j = 0; j < 4; ++j)
        acc[i][j].v = wmma_bf16(afr[i], bfr[j], acc[i][j].v);
  }

  // Epilogue: bias add + bf16 convert + store (D layout: VGPR r <-> M=r+8g, N=hl)
#pragma unroll
  for (int j = 0; j < 4; ++j) {
    const int e = col0 + j * 16 + hl;
    const float bb = bias[e];
#pragma unroll
    for (int i = 0; i < 4; ++i) {
      if (proj < 2) {
        __bf16* dst = (proj == 0) ? qo : ko;
#pragma unroll
        for (int r = 0; r < 8; ++r) {
          const int row = row0 + i * 16 + r + 8 * g;
          dst[(size_t)row * E_DIM + e] = (__bf16)(acc[i][j].e[r] + bb);
        }
      } else {
        // transposed store: vT[e, token]; 8 consecutive tokens per lane -> b128
        BF16x8 pk;
#pragma unroll
        for (int r = 0; r < 8; ++r) pk.e[r] = (__bf16)(acc[i][j].e[r] + bb);
        *(v8bf*)&vTo[(size_t)e * N_TOK + row0 + i * 16 + 8 * g] = pk.v;
      }
    }
  }
}

// =====================================================================
// Kernel 2: softmax stats. Per block: 64 query rows, 4 waves (16 rows each),
// stream key tiles of 64, online rowmax/rowsumexp. grid = N/64, block = 128.
// q/k chunks staged with CDNA5 async global->LDS copies (64-wide K chunk).
// =====================================================================
__global__ __launch_bounds__(128) void sdpa_stats_kernel(
    const __bf16* __restrict__ q, const __bf16* __restrict__ kmat,
    float* __restrict__ mrow, float* __restrict__ linv) {
  __shared__ __align__(16) __bf16 sq[64 * 64];
  __shared__ __align__(16) __bf16 sk[64 * 64];
  const int lane = threadIdx.x & 31;
  const int wave = threadIdx.x >> 5;           // 0..3 -> rows wave*16..+15
  const int g = lane >> 4, hl = lane & 15;
  const int row0 = blockIdx.x * 64;

  float mr[8], lr[8];
#pragma unroll
  for (int r = 0; r < 8; ++r) { mr[r] = -1e30f; lr[r] = 0.0f; }

  const int rr = threadIdx.x >> 1;             // 0..63
  const int cbase = (threadIdx.x & 1) * 32;    // 0 | 32

  for (int key0 = 0; key0 < N_TOK; key0 += 64) {
    if (key0 + 64 < N_TOK)
      __builtin_prefetch(&kmat[(size_t)(key0 + 64 + rr) * E_DIM], 0, 1);
    F32x8 s[4];
#pragma unroll
    for (int nt = 0; nt < 4; ++nt)
#pragma unroll
      for (int r = 0; r < 8; ++r) s[nt].e[r] = 0.0f;

    for (int k0 = 0; k0 < E_DIM; k0 += 64) {
      __syncthreads();  // previous chunk fully consumed
#pragma unroll
      for (int u = 0; u < 4; ++u) {
        const int c = cbase + u * 8;
        async_load_b128(lds_off(&sq[rr * 64 + c]),
                        &q[(size_t)(row0 + rr) * E_DIM + k0 + c]);
        async_load_b128(lds_off(&sk[rr * 64 + c]),
                        &kmat[(size_t)(key0 + rr) * E_DIM + k0 + c]);
      }
      wait_async0();
      __syncthreads();
#pragma unroll
      for (int kk = 0; kk < 2; ++kk) {
        const v16bf a = load_a_bf16(sq, 64, wave * 16, kk * 32, g, hl);
#pragma unroll
        for (int nt = 0; nt < 4; ++nt) {
          const v16bf b = load_b_bf16(sk, 64, nt * 16, kk * 32, g, hl);
          s[nt].v = wmma_bf16(a, b, s[nt].v);
        }
      }
    }

    // online stats per row (row M = r + 8g; reduce across the 16-lane group)
#pragma unroll
    for (int r = 0; r < 8; ++r) {
      float mx = -1e30f;
#pragma unroll
      for (int nt = 0; nt < 4; ++nt) mx = fmaxf(mx, s[nt].e[r]);
      mx *= SCALE;
#pragma unroll
      for (int off = 1; off < 16; off <<= 1) mx = fmaxf(mx, __shfl_xor(mx, off, 32));
      const float mnew = fmaxf(mr[r], mx);
      float sum = 0.0f;
#pragma unroll
      for (int nt = 0; nt < 4; ++nt) sum += __expf(s[nt].e[r] * SCALE - mnew);
#pragma unroll
      for (int off = 1; off < 16; off <<= 1) sum += __shfl_xor(sum, off, 32);
      lr[r] = lr[r] * __expf(mr[r] - mnew) + sum;
      mr[r] = mnew;
    }
  }

  if (hl == 0) {
#pragma unroll
    for (int r = 0; r < 8; ++r) {
      const int row = row0 + wave * 16 + r + 8 * g;
      mrow[row] = mr[r];
      linv[row] = 1.0f / lr[r];
    }
  }
}

// =====================================================================
// Kernel 3: O = softmax(qk^T) @ v. Per block: 32 query rows x full E,
// 8 waves: wave = (rg, es): rows rg*16..+15, cols es*256..+255.
// S recompute per key tile via WMMA (async-staged q/k chunks);
// P through LDS (D->A layout); P@V B-frags from L2-resident vT.
// grid = N/32, block = 256.
// =====================================================================
__global__ __launch_bounds__(256) void sdpa_out_kernel(
    const __bf16* __restrict__ q, const __bf16* __restrict__ kmat,
    const __bf16* __restrict__ vT, const float* __restrict__ mrow,
    const float* __restrict__ linv, float* __restrict__ out) {
  __shared__ __align__(16) __bf16 sq[32 * 64];
  __shared__ __align__(16) __bf16 sk[64 * 64];
  __shared__ __align__(16) __bf16 sp[32 * 64];
  const int lane = threadIdx.x & 31;
  const int wave = threadIdx.x >> 5;
  const int g = lane >> 4, hl = lane & 15;
  const int rg = wave >> 2;   // row group (16 rows)
  const int es = wave & 3;    // e-slice (256 cols) == this wave's S n-tile
  const int row0 = blockIdx.x * 32;

  float mr[8], li[8];
#pragma unroll
  for (int r = 0; r < 8; ++r) {
    const int row = row0 + rg * 16 + r + 8 * g;
    mr[r] = mrow[row];
    li[r] = linv[row];
  }

  F32x8 acc[16];
#pragma unroll
  for (int j = 0; j < 16; ++j)
#pragma unroll
    for (int r = 0; r < 8; ++r) acc[j].e[r] = 0.0f;

  const int t = threadIdx.x;
  const int qr = t >> 3, qc = (t & 7) * 8;   // sq: 32x64, one b128 per thread
  const int kr = t >> 2, kc = (t & 3) * 16;  // sk: 64x64, two b128 per thread

  for (int key0 = 0; key0 < N_TOK; key0 += 64) {
    if (key0 + 64 < N_TOK)
      __builtin_prefetch(&kmat[(size_t)(key0 + 64 + kr) * E_DIM], 0, 1);

    // ---- Phase A: S tile (rg, es) = q[rg rows] . k[es*16 keys]^T over E ----
    F32x8 s;
#pragma unroll
    for (int r = 0; r < 8; ++r) s.e[r] = 0.0f;
    for (int k0 = 0; k0 < E_DIM; k0 += 64) {
      __syncthreads();  // previous chunk fully consumed
      async_load_b128(lds_off(&sq[qr * 64 + qc]),
                      &q[(size_t)(row0 + qr) * E_DIM + k0 + qc]);
      async_load_b128(lds_off(&sk[kr * 64 + kc]),
                      &kmat[(size_t)(key0 + kr) * E_DIM + k0 + kc]);
      async_load_b128(lds_off(&sk[kr * 64 + kc + 8]),
                      &kmat[(size_t)(key0 + kr) * E_DIM + k0 + kc + 8]);
      wait_async0();
      __syncthreads();
#pragma unroll
      for (int kk = 0; kk < 2; ++kk) {
        const v16bf a = load_a_bf16(sq, 64, rg * 16, kk * 32, g, hl);
        const v16bf b = load_b_bf16(sk, 64, es * 16, kk * 32, g, hl);
        s.v = wmma_bf16(a, b, s.v);
      }
    }

    // ---- softmax transform, D-layout -> LDS (A-layout readable) ----
#pragma unroll
    for (int r = 0; r < 8; ++r) {
      const float p = __expf(s.e[r] * SCALE - mr[r]) * li[r];
      sp[(rg * 16 + r + 8 * g) * 64 + es * 16 + hl] = (__bf16)p;
    }
    __syncthreads();

    // ---- Phase B: acc += P(16x64) @ V(64 x 256-slice), B from vT ----
#pragma unroll
    for (int kk = 0; kk < 2; ++kk) {
      const v16bf a = load_a_bf16(sp, 64, rg * 16, kk * 32, g, hl);
#pragma unroll
      for (int j = 0; j < 16; ++j) {
        const int e0 = es * 256 + j * 16;
        const v16bf b = load_b_bf16(vT, N_TOK, e0, key0 + kk * 32, g, hl);
        acc[j].v = wmma_bf16(a, b, acc[j].v);
      }
    }
  }

  // ---- store O (f32) ----
#pragma unroll
  for (int j = 0; j < 16; ++j) {
    const int e = es * 256 + j * 16 + hl;
#pragma unroll
    for (int r = 0; r < 8; ++r) {
      const int row = row0 + rg * 16 + r + 8 * g;
      out[(size_t)row * E_DIM + e] = acc[j].e[r];
    }
  }
}

// =====================================================================
extern "C" void kernel_launch(void* const* d_in, const int* in_sizes, int n_in,
                              void* d_out, int out_size, void* d_ws, size_t ws_size,
                              hipStream_t stream) {
  const float* x  = (const float*)d_in[0];
  const float* Wq = (const float*)d_in[1];
  const float* bq = (const float*)d_in[2];
  const float* Wk = (const float*)d_in[3];
  const float* bk = (const float*)d_in[4];
  const float* Wv = (const float*)d_in[5];
  const float* bv = (const float*)d_in[6];
  float* out = (float*)d_out;

  // workspace layout: q | k | vT (bf16), then m, 1/l (f32). ~48.1 MB total.
  const size_t NE = (size_t)N_TOK * E_DIM;
  __bf16* qb  = (__bf16*)d_ws;
  __bf16* kb  = qb + NE;
  __bf16* vTb = kb + NE;
  float*  mr  = (float*)(vTb + NE);
  float*  li  = mr + N_TOK;

  dim3 gproj(N_TOK / 128, E_DIM / 256, 3);
  sdpa_qkv_proj_kernel<<<gproj, 256, 0, stream>>>(x, Wq, bq, Wk, bk, Wv, bv,
                                                  qb, kb, vTb);
  sdpa_stats_kernel<<<dim3(N_TOK / 64), 128, 0, stream>>>(qb, kb, mr, li);
  sdpa_out_kernel<<<dim3(N_TOK / 32), 256, 0, stream>>>(qb, kb, vTb, mr, li, out);
}
